// EdgeNetwork_34196529611289
// MI455X (gfx1250) — matrix operand order repaired and checked by reading, not compiled, not run on hardware
//
#include <hip/hip_runtime.h>

typedef __attribute__((ext_vector_type(4)))  _Float16 v4h;
typedef __attribute__((ext_vector_type(8)))  _Float16 v8h;
typedef __attribute__((ext_vector_type(16))) _Float16 v16h;
typedef __attribute__((ext_vector_type(8)))  float    v8f;

#define N_EDGES    640000
#define N_NODES    100000
#define NF         128      // node feature width
#define MIDF       192      // hidden width
#define OUTF       128      // output width
#define K1         256      // GEMM1 K (2*NF); dR column is a rank-1 epilogue
#define TM         128      // edges per workgroup
#define NTHREADS   512      // 16 wave32 -> 4 waves/SIMD
#define AIN_STRIDE 264      // 256 + 8 halves pad (stride%64dw == 4 banks -> conflict free)
#define H_STRIDE   200      // 192 + 8 halves pad

// ---- helpers -------------------------------------------------------------

__device__ __forceinline__ v16h ld16(const _Float16* p0, const _Float16* p1) {
  v8h a = *(const v8h*)p0;
  v8h b = *(const v8h*)p1;
  return __builtin_shufflevector(a, b, 0,1,2,3,4,5,6,7,8,9,10,11,12,13,14,15);
}

__device__ __forceinline__ v8f wmma16(v16h a, v16h b, v8f c) {
  // D = A(16xK32) * B(K32x16) + C, f32 accumulate
  return __builtin_amdgcn_wmma_f32_16x16x32_f16(false, a, false, b, (short)0, c,
                                                false, false);
}

// ---- prologue: f32 -> f16 conversions ------------------------------------

__global__ void cvt_h_kernel(const float* __restrict__ h, _Float16* __restrict__ hf,
                             int n4) {
  int i = blockIdx.x * blockDim.x + threadIdx.x;
  if (i < n4) {
    float4 v = ((const float4*)h)[i];
    v4h o;
    o[0] = (_Float16)v.x; o[1] = (_Float16)v.y;
    o[2] = (_Float16)v.z; o[3] = (_Float16)v.w;
    ((v4h*)hf)[i] = o;
  }
}

__global__ void cvt_w_kernel(const float* __restrict__ W1, const float* __restrict__ W2,
                             _Float16* __restrict__ W1T, _Float16* __restrict__ W2T) {
  int i = blockIdx.x * blockDim.x + threadIdx.x;
  if (i < MIDF * K1) {                 // W1T[n][k] = W1[k][n], k < 256
    int n = i / K1, k = i % K1;
    W1T[i] = (_Float16)W1[k * MIDF + n];
  }
  if (i < OUTF * MIDF) {               // W2T[o][k] = W2[k][o]
    int o = i / MIDF, k = i % MIDF;
    W2T[i] = (_Float16)W2[k * OUTF + o];
  }
}

// ---- main fused edge-MLP kernel ------------------------------------------

__global__ __launch_bounds__(NTHREADS) void edge_mlp_kernel(
    const _Float16* __restrict__ hf,       // [N_NODES][NF] f16
    const float*    __restrict__ dR,       // [E]
    const int*      __restrict__ src_idx,  // [E]
    const int*      __restrict__ dst_idx,  // [E]
    const _Float16* __restrict__ W1T,      // [MIDF][K1] f16 (transposed)
    const float*    __restrict__ w1last,   // W1 row 256: [MIDF] f32
    const float*    __restrict__ b1,       // [MIDF]
    const _Float16* __restrict__ W2T,      // [OUTF][MIDF] f16 (transposed)
    const float*    __restrict__ b2,       // [OUTF]
    float*          __restrict__ out)      // [E][OUTF]
{
  __shared__ _Float16 Ain[TM * AIN_STRIDE];   // inp[e][k] f16  (dst | src)
  __shared__ _Float16 Hs[TM * H_STRIDE];      // hid[e][n] f16
  __shared__ float    sdR[TM];

  const int tid = threadIdx.x;
  const int e0  = blockIdx.x * TM;

  // -------- Phase A: gather 128 edges (4 threads per edge: dst/src x 2 halves)
  {
    const int r   = tid >> 2;        // 0..127  edge within tile
    const int sub = tid & 3;
    const int s   = sub >> 1;        // 0 = dst (k 0..127), 1 = src (k 128..255)
    const int q   = sub & 1;         // 64-half sub-segment
    const int node = s ? src_idx[e0 + r] : dst_idx[e0 + r];
    const float4* sp = (const float4*)(hf + (size_t)node * NF + q * 64);
    float4* dp = (float4*)(Ain + r * AIN_STRIDE + s * NF + q * 64);
#pragma unroll
    for (int j = 0; j < 8; ++j) dp[j] = sp[j];
    if (sub == 0) sdR[r] = dR[e0 + r];
  }

  const int wave  = tid >> 5;        // 0..15
  const int lo    = tid & 15;        // column (edge) within tile
  const int hi    = (tid >> 4) & 1;  // half-wave selector
  const int etile = wave >> 1;       // pair of waves per edge tile
  const int grp   = wave & 1;        // which half of the n/o tiles

  const _Float16* brow   = Ain + (etile * 16 + lo) * AIN_STRIDE;
  _Float16*       hrow   = Hs  + (etile * 16 + lo) * H_STRIDE;
  const _Float16* abase1 = W1T + ((grp * 6) * 16 + lo) * K1 + hi * 8;
  const _Float16* abase2 = W2T + ((grp * 4) * 16 + lo) * MIDF + hi * 8;

  auto ldA1 = [&](int i) -> v16h {   // i = ks*6 + j
    const int ks = i / 6, j = i % 6;
    const _Float16* p = abase1 + j * (16 * K1) + ks * 32;
    return ld16(p, p + 16);          // K {hi*8..+7, 16+hi*8..+7}
  };
  auto ldB1 = [&](int ks) -> v16h {
    const _Float16* p = brow + ks * 32 + hi * 8;
    return ld16(p, p + 16);          // K {hi*8..+7, 16+hi*8..+7}
  };

  // prefetch first two weight tiles while waiting on the gather barrier
  v16h Aa = ldA1(0);
  v16h Ab = ldA1(1);

  __syncthreads();

  // -------- GEMM1: hidT[6x16 n x 16e] = W1T x inpT   (software pipelined)
  v8f acc[6];
#pragma unroll
  for (int j = 0; j < 6; ++j)
#pragma unroll
    for (int v = 0; v < 8; ++v) acc[j][v] = 0.0f;

  v16h B0 = ldB1(0);
  v16h B1 = ldB1(1);

#pragma unroll
  for (int i = 0; i < 48; ++i) {
    const int ks = i / 6, j = i % 6;
    v16h An = Aa;
    if (i + 2 < 48) An = ldA1(i + 2);      // depth-2 A prefetch
    acc[j] = wmma16(Aa, B0, acc[j]);
    if (j == 5) {                          // rotate B at k-group boundary
      B0 = B1;
      if (ks + 2 < 8) B1 = ldB1(ks + 2);
    }
    Aa = Ab; Ab = An;
  }

  // epilogue 1: + dR*W1[256,:] + b1, relu, f16-pack into Hs[e][n]
  const float dr = sdR[etile * 16 + lo];
#pragma unroll
  for (int j = 0; j < 6; ++j) {
    const int n0 = (grp * 6 + j) * 16 + hi * 8;    // 8 contiguous n per lane
    float4 wa = *(const float4*)(w1last + n0);
    float4 wb = *(const float4*)(w1last + n0 + 4);
    float4 ba = *(const float4*)(b1 + n0);
    float4 bb = *(const float4*)(b1 + n0 + 4);
    float wv[8] = {wa.x, wa.y, wa.z, wa.w, wb.x, wb.y, wb.z, wb.w};
    float bv[8] = {ba.x, ba.y, ba.z, ba.w, bb.x, bb.y, bb.z, bb.w};
    v8h hp;
#pragma unroll
    for (int v = 0; v < 8; ++v) {
      float x = acc[j][v] + dr * wv[v] + bv[v];
      hp[v] = (_Float16)(x > 0.0f ? x : 0.0f);
    }
    *(v8h*)(hrow + n0) = hp;                       // one ds_store_b128
  }

  auto ldA2 = [&](int i) -> v16h {   // i = ks*4 + j
    const int ks = i >> 2, j = i & 3;
    const _Float16* p = abase2 + j * (16 * MIDF) + ks * 32;
    return ld16(p, p + 16);
  };
  auto ldB2 = [&](int ks) -> v16h {
    const _Float16* p = hrow + ks * 32 + hi * 8;
    return ld16(p, p + 16);
  };

  // prefetch W2T tiles under the barrier (global, independent of Hs)
  Aa = ldA2(0);
  Ab = ldA2(1);

  // each edge tile's Hs rows are written by a PAIR of waves -> barrier
  __syncthreads();

  // -------- GEMM2: outT[4x16 o x 16e] = W2T x hidT   (software pipelined)
  v8f acc2[4];
#pragma unroll
  for (int j = 0; j < 4; ++j)
#pragma unroll
    for (int v = 0; v < 8; ++v) acc2[j][v] = 0.0f;

  B0 = ldB2(0);
  B1 = ldB2(1);

#pragma unroll
  for (int i = 0; i < 24; ++i) {
    const int ks = i >> 2, j = i & 3;
    v16h An = Aa;
    if (i + 2 < 24) An = ldA2(i + 2);
    acc2[j] = wmma16(Aa, B0, acc2[j]);
    if (j == 3) {
      B0 = B1;
      if (ks + 2 < 6) B1 = ldB2(ks + 2);
    }
    Aa = Ab; Ab = An;
  }

  // epilogue 2: + b2, tanh, store f32 rows of out[e][:]
  float* orow = out + (size_t)(e0 + etile * 16 + lo) * OUTF;
#pragma unroll
  for (int j = 0; j < 4; ++j) {
    const int o0 = (grp * 4 + j) * 16 + hi * 8;
    float4 ba = *(const float4*)(b2 + o0);
    float4 bb = *(const float4*)(b2 + o0 + 4);
    float bv[8] = {ba.x, ba.y, ba.z, ba.w, bb.x, bb.y, bb.z, bb.w};
    float y[8];
#pragma unroll
    for (int v = 0; v < 8; ++v) y[v] = tanhf(acc2[j][v] + bv[v]);
    *(float4*)(orow + o0)     = make_float4(y[0], y[1], y[2], y[3]);
    *(float4*)(orow + o0 + 4) = make_float4(y[4], y[5], y[6], y[7]);
  }
}

// ---- launcher -------------------------------------------------------------

extern "C" void kernel_launch(void* const* d_in, const int* in_sizes, int n_in,
                              void* d_out, int out_size, void* d_ws, size_t ws_size,
                              hipStream_t stream) {
  (void)in_sizes; (void)n_in; (void)out_size; (void)ws_size;

  const float* h   = (const float*)d_in[0];
  const float* dR  = (const float*)d_in[1];
  const int*   src = (const int*)d_in[2];
  const int*   dst = (const int*)d_in[3];
  const float* W1  = (const float*)d_in[4];
  const float* b1  = (const float*)d_in[5];
  const float* W2  = (const float*)d_in[6];
  const float* b2  = (const float*)d_in[7];

  // workspace layout (all 256B aligned)
  const size_t HF_BYTES  = (size_t)N_NODES * NF * 2;        // 25,600,000
  const size_t W1T_BYTES = (size_t)MIDF * K1 * 2;           // 98,304
  _Float16* hf16 = (_Float16*)d_ws;
  _Float16* W1T  = (_Float16*)((char*)d_ws + HF_BYTES);
  _Float16* W2T  = (_Float16*)((char*)d_ws + HF_BYTES + W1T_BYTES);

  // prologue conversions
  const int n4 = N_NODES * NF / 4;                          // 3,200,000 float4s
  cvt_h_kernel<<<(n4 + 255) / 256, 256, 0, stream>>>(h, hf16, n4);
  cvt_w_kernel<<<(MIDF * K1 + 255) / 256, 256, 0, stream>>>(W1, W2, W1T, W2T);

  // fused edge MLP: 640000 / 128 = 5000 workgroups of 16 wave32
  edge_mlp_kernel<<<N_EDGES / TM, NTHREADS, 0, stream>>>(
      hf16, dR, src, dst, W1T, W1 + (size_t)K1 * MIDF, b1, W2T, b2, (float*)d_out);
}
